// SwinTransformerBlock_4904852652200
// MI455X (gfx1250) — matrix operand order repaired and checked by reading, not compiled, not run on hardware
//
#include <hip/hip_runtime.h>
#include <math.h>

// ---------------------------------------------------------------------------
// Problem constants
// ---------------------------------------------------------------------------
#define BATCH   16
#define SEQ     4096
#define CDIM    512
#define HEADS   8
#define HDIM    64
#define WIN     64
#define SHIFT_N 32
#define HIDDEN  2048
#define NWIN    (BATCH * SEQ / WIN)       // 1024 windows
#define MROWS   (BATCH * SEQ)             // 65536 token rows
#define LDW     72                        // padded LDS row (bf16 elems)

typedef __attribute__((ext_vector_type(16))) __bf16 v16bf;
typedef __attribute__((ext_vector_type(8)))  float  v8f;

union BF16x16 { v16bf v; uint4 q[2]; };

// gfx1250 async global->LDS path (ASYNCcnt-tracked), guarded by __has_builtin
#if defined(__has_builtin)
#  if __has_builtin(__builtin_amdgcn_global_load_async_to_lds_b128)
#    define ASYNC_LDS 1
#  endif
#endif

// exact pointee type from the builtin's prototype: vector_size(16) int
typedef int v4i_vs __attribute__((vector_size(16)));

__device__ __forceinline__ void copy16_g2l_async(unsigned short* lds,
                                                 const unsigned short* g) {
#if defined(ASYNC_LDS)
  __builtin_amdgcn_global_load_async_to_lds_b128(
      (__attribute__((address_space(1))) v4i_vs*)(uintptr_t)g,
      (__attribute__((address_space(3))) v4i_vs*)lds, 0, 0);
#else
  *(uint4*)lds = *(const uint4*)g;
#endif
}

__device__ __forceinline__ void wait_async_all() {
#if defined(ASYNC_LDS)
#  if __has_builtin(__builtin_amdgcn_s_wait_asynccnt)
  __builtin_amdgcn_s_wait_asynccnt(0);
#  else
  asm volatile("s_wait_asynccnt 0x0" ::: "memory");
#  endif
#endif
}

__device__ __forceinline__ unsigned short f2bf(float f) {
  unsigned int u = __float_as_uint(f);
  u += 0x7FFFu + ((u >> 16) & 1u);                 // round-to-nearest-even
  return (unsigned short)(u >> 16);
}
__device__ __forceinline__ float bf2f(unsigned short h) {
  return __uint_as_float(((unsigned int)h) << 16);
}

__device__ __forceinline__ v8f wmma_bf16(v16bf a, v16bf b, v8f c) {
  return __builtin_amdgcn_wmma_f32_16x16x32_bf16(
      /*neg_a=*/false, a, /*neg_b=*/false, b,
      /*c_mod=*/(short)0, c, /*reuse_a=*/false, /*reuse_b=*/false);
}

// A-frag (16x32 bf16, M x K): lane L<16 holds row M=L, K = kc+[0..15];
// lane L>=16 holds row M=L-16, K = kc+[16..31].
__device__ __forceinline__ v16bf load_a_frag(const unsigned short* base, int kc, int lane) {
  int row = lane & 15;
  int ko  = kc + ((lane >> 4) << 4);
  BF16x16 u;
  u.q[0] = *(const uint4*)(base + row * LDW + ko);
  u.q[1] = *(const uint4*)(base + row * LDW + ko + 8);
  return u.v;
}
// B-frag (32x16 bf16, K x N) read from row-major [N][K] storage:
// lane L<16 holds col N=L, K = kc+[0..15]; lane L>=16 holds col N=L-16, K = kc+[16..31].
__device__ __forceinline__ v16bf load_b_frag(const unsigned short* base, int kc, int lane) {
  int n  = lane & 15;
  int ko = kc + ((lane >> 4) << 4);
  BF16x16 u;
  u.q[0] = *(const uint4*)(base + n * LDW + ko);
  u.q[1] = *(const uint4*)(base + n * LDW + ko + 8);
  return u.v;
}

// ---------------------------------------------------------------------------
// Elementwise helpers
// ---------------------------------------------------------------------------
__global__ void f2bf_kernel(const float* __restrict__ src,
                            unsigned short* __restrict__ dst, int n) {
  int id = blockIdx.x * 256 + threadIdx.x;
  if (id < n) dst[id] = f2bf(src[id]);
}

// shifted_x = roll(x, -SHIFT) with tail zeroed, converted to bf16 (window order)
__global__ void prep_x_kernel(const float* __restrict__ x,
                              unsigned short* __restrict__ xw) {
  size_t id = (size_t)blockIdx.x * 256 + threadIdx.x;   // 33.5M
  int m  = (int)(id >> 9);        // token row b*SEQ + l'
  int lp = m & (SEQ - 1);
  float v = (lp < SEQ - SHIFT_N) ? x[id + (size_t)SHIFT_N * CDIM] : 0.f;
  xw[id] = f2bf(v);
}

// mw = shifted mask as float; new_mask output = mask with first SHIFT rows false
__global__ void masks_kernel(const unsigned char* __restrict__ mask,
                             float* __restrict__ mw,
                             float* __restrict__ nm_out) {
  int id = blockIdx.x * 256 + threadIdx.x;   // 65536
  int l  = id & (SEQ - 1);
  float sm = (l < SEQ - SHIFT_N) ? (mask[id + SHIFT_N] ? 1.f : 0.f) : 0.f;
  mw[id] = sm;
  nm_out[id] = (l < SHIFT_N) ? 0.f : (mask[id] ? 1.f : 0.f);
}

// CPB: table[i][h] = relu(t_i * w1 + b1) . w2[h]
__global__ void cpb_table_kernel(const float* __restrict__ w1,
                                 const float* __restrict__ b1,
                                 const float* __restrict__ w2,
                                 float* __restrict__ table) {
  int id = blockIdx.x * 128 + threadIdx.x;
  if (id >= 127 * HEADS) return;
  int i = id >> 3, h = id & 7;
  float t  = (float)(i - (WIN - 1)) / (float)(WIN - 1) * 8.f;
  float tt = copysignf(log2f(fabsf(t) + 1.f) / 3.f, t);   // log2(8)=3
  float acc = 0.f;
  for (int j = 0; j < 512; ++j) {
    float hj = fmaxf(tt * w1[j] + b1[j], 0.f);
    acc += hj * w2[h * 512 + j];
  }
  table[i * HEADS + h] = acc;
}

// bias[h][i][j] = 16*sigmoid(table[i-j+63][h])
__global__ void cpb_bias_kernel(const float* __restrict__ table,
                                float* __restrict__ bias) {
  int id = blockIdx.x * 256 + threadIdx.x;   // 8*64*64 = 32768
  int h = id >> 12, rem = id & 4095, i = rem >> 6, j = rem & 63;
  float tv = table[(i - j + WIN - 1) * HEADS + h];
  bias[id] = 16.f / (1.f + __expf(-tv));
}

// ---------------------------------------------------------------------------
// Generic tiled WMMA GEMM: C = act(A[M,K] @ W[N,K]^T + bias)
//   mode 0: QKV -> scatter per-head q/k/v bf16 tiles (biasA=q_bias, biasB=v_bias)
//   mode 1: f32 row-major output (+biasA)
//   mode 2: GELU(.) -> bf16 row-major output (+biasA)
// block: 256 threads = 8 waves; tile 128(M) x 64(N); K staged 64/iter in LDS
// via async global->LDS copies (ASYNCcnt) with next-slice prefetch.
// ---------------------------------------------------------------------------
__global__ __launch_bounds__(256) void gemm_kernel(
    const unsigned short* __restrict__ A,
    const unsigned short* __restrict__ Wt,
    const float* __restrict__ biasA,
    const float* __restrict__ biasB,
    int M, int N, int K, int mode,
    unsigned short* __restrict__ ob0,
    unsigned short* __restrict__ ob1,
    unsigned short* __restrict__ ob2,
    float* __restrict__ of) {
  __shared__ alignas(16) unsigned short As[128 * LDW];
  __shared__ alignas(16) unsigned short Bs[64 * LDW];
  int tid  = threadIdx.x;
  int lane = tid & 31, wid = tid >> 5;
  int wm = wid & 3, wn = wid >> 2;          // 4 x 2 wave grid
  int m0 = blockIdx.y << 7;
  int n0 = blockIdx.x << 6;

  v8f z = {0.f, 0.f, 0.f, 0.f, 0.f, 0.f, 0.f, 0.f};
  v8f acc[2][2] = {{z, z}, {z, z}};

  for (int k0 = 0; k0 < K; k0 += 64) {
#if defined(ASYNC_LDS)
    // issue all tile copies asynchronously (no VGPR round-trip)
#pragma unroll
    for (int j = 0; j < 4; ++j) {             // A tile: 128 x 64
      int i = tid + j * 256;
      int row = i >> 3, c8 = (i & 7) << 3;
      copy16_g2l_async(&As[row * LDW + c8],
                       &A[(size_t)(m0 + row) * K + k0 + c8]);
    }
#pragma unroll
    for (int j = 0; j < 2; ++j) {             // W tile: 64 x 64 ([N][K])
      int i = tid + j * 256;
      int row = i >> 3, c8 = (i & 7) << 3;
      copy16_g2l_async(&Bs[row * LDW + c8],
                       &Wt[(size_t)(n0 + row) * K + k0 + c8]);
    }
#else
    // fallback: batch loads into registers, then store (keeps loads in flight)
    uint4 ra[4], rb[2];
#pragma unroll
    for (int j = 0; j < 4; ++j) {
      int i = tid + j * 256;
      int row = i >> 3, c8 = (i & 7) << 3;
      ra[j] = *(const uint4*)&A[(size_t)(m0 + row) * K + k0 + c8];
    }
#pragma unroll
    for (int j = 0; j < 2; ++j) {
      int i = tid + j * 256;
      int row = i >> 3, c8 = (i & 7) << 3;
      rb[j] = *(const uint4*)&Wt[(size_t)(n0 + row) * K + k0 + c8];
    }
#pragma unroll
    for (int j = 0; j < 4; ++j) {
      int i = tid + j * 256;
      int row = i >> 3, c8 = (i & 7) << 3;
      *(uint4*)&As[row * LDW + c8] = ra[j];
    }
#pragma unroll
    for (int j = 0; j < 2; ++j) {
      int i = tid + j * 256;
      int row = i >> 3, c8 = (i & 7) << 3;
      *(uint4*)&Bs[row * LDW + c8] = rb[j];
    }
#endif
    // prefetch next K-slice into cache while current tile is copied/consumed
    if (k0 + 64 < K) {
      int rowp = tid & 127, cp = (tid >> 7) << 5;   // 128 rows x 2 half-lines
      __builtin_prefetch(&A[(size_t)(m0 + rowp) * K + k0 + 64 + cp], 0, 1);
      if (tid < 128) {
        int rowb = tid & 63, cb = (tid >> 6) << 5;
        __builtin_prefetch(&Wt[(size_t)(n0 + rowb) * K + k0 + 64 + cb], 0, 1);
      }
    }
    wait_async_all();
    __syncthreads();
#pragma unroll
    for (int kc = 0; kc < 64; kc += 32) {
      v16bf a0 = load_a_frag(&As[(wm * 32) * LDW],      kc, lane);
      v16bf a1 = load_a_frag(&As[(wm * 32 + 16) * LDW], kc, lane);
      v16bf b0 = load_b_frag(&Bs[(wn * 32) * LDW],      kc, lane);
      v16bf b1 = load_b_frag(&Bs[(wn * 32 + 16) * LDW], kc, lane);
      acc[0][0] = wmma_bf16(a0, b0, acc[0][0]);
      acc[0][1] = wmma_bf16(a0, b1, acc[0][1]);
      acc[1][0] = wmma_bf16(a1, b0, acc[1][0]);
      acc[1][1] = wmma_bf16(a1, b1, acc[1][1]);
    }
    __syncthreads();
  }

  // Epilogue: D-layout lane l, reg r -> M = r + 8*(l>>4), N = l&15
#pragma unroll
  for (int tm = 0; tm < 2; ++tm)
#pragma unroll
    for (int tn = 0; tn < 2; ++tn) {
      int mBase = m0 + wm * 32 + tm * 16 + ((lane >> 4) << 3);
      int nn    = n0 + wn * 32 + tn * 16 + (lane & 15);
#pragma unroll
      for (int r = 0; r < 8; ++r) {
        int m   = mBase + r;
        float v = acc[tm][tn][r];
        if (mode == 0) {
          int which = nn >> 9;                           // 0=q 1=k 2=v
          float bi = (which == 0) ? biasA[nn]
                     : (which == 2 ? biasB[nn & 511] : 0.f);
          v += bi;
          int h = (nn >> 6) & 7, dd = nn & 63;
          int wdw = m >> 6, tok = m & 63;
          size_t idx = ((((size_t)wdw * HEADS) + h) * WIN + tok) * HDIM + dd;
          unsigned short pv = f2bf(v);
          if (which == 0) ob0[idx] = pv;
          else if (which == 1) ob1[idx] = pv;
          else ob2[idx] = pv;
        } else if (mode == 1) {
          of[(size_t)m * N + nn] = v + biasA[nn];
        } else {
          float xv = v + biasA[nn];
          float gl = 0.5f * xv * (1.f + erff(xv * 0.70710678118654752f));
          ob0[(size_t)m * N + nn] = f2bf(gl);
        }
      }
    }
}

// ---------------------------------------------------------------------------
// Attention: one block per (window, head), 128 threads (4 waves).
// Cosine attn: normalize q,k rows; scale folded into q; WMMA qk^T;
// softmax(+CPB bias+shift mask); WMMA P@V; write bf16 [token][h*64+d].
// ---------------------------------------------------------------------------
__global__ __launch_bounds__(128) void attn_kernel(
    const unsigned short* __restrict__ qb,
    const unsigned short* __restrict__ kb,
    const unsigned short* __restrict__ vb,
    const float* __restrict__ bias,          // [H][64][64]
    const float* __restrict__ mw,            // [NWIN*64]
    const float* __restrict__ logit_scale,   // [H]
    unsigned short* __restrict__ attn_out) { // [MROWS][CDIM] bf16
  __shared__ alignas(16) unsigned short qs[64 * LDW];   // qn*scale, reused for P
  __shared__ alignas(16) unsigned short ks[64 * LDW];   // kn (row-major [m][d])
  __shared__ alignas(16) unsigned short vsT[64 * LDW];  // v transposed [d][m]
  __shared__ float ss[64 * 68];
  int tid = threadIdx.x;
  int wh = blockIdx.x;
  int w = wh >> 3, h = wh & 7;
  size_t base = (size_t)wh * WIN * HDIM;
  const unsigned short* qp = qb + base;
  const unsigned short* kp = kb + base;
  const unsigned short* vp = vb + base;
  float scale = __expf(fminf(logit_scale[h], 4.6051701859880914f)); // ln(100)

  if (tid < 64) {
    int r = tid;
    float s2 = 0.f;
    for (int d = 0; d < 64; ++d) { float f = bf2f(qp[r * 64 + d]); s2 += f * f; }
    float inv = scale / fmaxf(sqrtf(s2), 1e-12f);
    for (int d = 0; d < 64; ++d) qs[r * LDW + d] = f2bf(bf2f(qp[r * 64 + d]) * inv);
    for (int d = 0; d < 64; ++d) vsT[d * LDW + r] = vp[r * 64 + d];
  } else {
    int r = tid - 64;
    float s2 = 0.f;
    for (int d = 0; d < 64; ++d) { float f = bf2f(kp[r * 64 + d]); s2 += f * f; }
    float inv = 1.f / fmaxf(sqrtf(s2), 1e-12f);
    for (int d = 0; d < 64; ++d) ks[r * LDW + d] = f2bf(bf2f(kp[r * 64 + d]) * inv);
  }
  __syncthreads();

  int lane = tid & 31, wid = tid >> 5;
  v8f z = {0.f, 0.f, 0.f, 0.f, 0.f, 0.f, 0.f, 0.f};
  // S = qn @ kn^T  (B[k=d][n=m] = kn[m][d] -> contiguous b-frag from ks)
  v8f s4[4] = {z, z, z, z};
  {
    v16bf a0 = load_a_frag(&qs[wid * 16 * LDW], 0, lane);
    v16bf a1 = load_a_frag(&qs[wid * 16 * LDW], 32, lane);
#pragma unroll
    for (int nt = 0; nt < 4; ++nt) {
      v16bf b0 = load_b_frag(&ks[nt * 16 * LDW], 0, lane);
      v16bf b1 = load_b_frag(&ks[nt * 16 * LDW], 32, lane);
      s4[nt] = wmma_bf16(a0, b0, s4[nt]);
      s4[nt] = wmma_bf16(a1, b1, s4[nt]);
    }
  }
  int row0 = wid * 16 + ((lane >> 4) << 3);
  int colL = lane & 15;
#pragma unroll
  for (int nt = 0; nt < 4; ++nt)
#pragma unroll
    for (int r = 0; r < 8; ++r)
      ss[(row0 + r) * 68 + nt * 16 + colL] = s4[nt][r];
  __syncthreads();

  if (tid < 64) {
    int r = tid;
    float mwr = mw[w * WIN + r];
    const float* bp = bias + (h * WIN + r) * WIN;
    float mx = -1e30f;
    for (int m = 0; m < 64; ++m) {
      float val = ss[r * 68 + m] + bp[m] + (1.f - mw[w * WIN + m] * mwr) * -10000.f;
      ss[r * 68 + m] = val;
      mx = fmaxf(mx, val);
    }
    float sum = 0.f;
    for (int m = 0; m < 64; ++m) {
      float e = __expf(ss[r * 68 + m] - mx);
      ss[r * 68 + m] = e;
      sum += e;
    }
    float inv = 1.f / sum;
    for (int m = 0; m < 64; ++m) qs[r * LDW + m] = f2bf(ss[r * 68 + m] * inv);
  }
  __syncthreads();

  // O = P @ V  (B[k=m][n=d] = vsT[d][m] -> contiguous b-frag from vsT)
  v8f o4[4] = {z, z, z, z};
  {
    v16bf p0 = load_a_frag(&qs[wid * 16 * LDW], 0, lane);
    v16bf p1 = load_a_frag(&qs[wid * 16 * LDW], 32, lane);
#pragma unroll
    for (int nt = 0; nt < 4; ++nt) {
      v16bf b0 = load_b_frag(&vsT[nt * 16 * LDW], 0, lane);
      v16bf b1 = load_b_frag(&vsT[nt * 16 * LDW], 32, lane);
      o4[nt] = wmma_bf16(p0, b0, o4[nt]);
      o4[nt] = wmma_bf16(p1, b1, o4[nt]);
    }
  }
#pragma unroll
  for (int nt = 0; nt < 4; ++nt)
#pragma unroll
    for (int r = 0; r < 8; ++r)
      attn_out[(size_t)(w * WIN + row0 + r) * CDIM + h * HDIM + nt * 16 + colL] =
          f2bf(o4[nt][r]);
}

// ---------------------------------------------------------------------------
// Residual + LayerNorm: out = shortcut + LN(src_row) * g + b
// shifted=1: src row = roll(+SHIFT) of window-order rows, first SHIFT rows zero.
// ---------------------------------------------------------------------------
__global__ __launch_bounds__(128) void ln_kernel(
    const float* __restrict__ src,
    const float* __restrict__ shortcut,
    const float* __restrict__ g,
    const float* __restrict__ bb,
    int shifted,
    float* __restrict__ out,
    unsigned short* __restrict__ out_bf) {
  __shared__ float rs[128], rs2[128];
  int row = blockIdx.x;
  int tid = threadIdx.x;
  int l = row & (SEQ - 1);
  bool zero = shifted && (l < SHIFT_N);
  const float* sp = src + (size_t)(shifted ? (row - SHIFT_N) : row) * CDIM;
  float vals[4];
  float s = 0.f, s2 = 0.f;
#pragma unroll
  for (int i = 0; i < 4; ++i) {
    int c = tid + i * 128;
    float v = zero ? 0.f : sp[c];
    vals[i] = v; s += v; s2 += v * v;
  }
  rs[tid] = s; rs2[tid] = s2;
  __syncthreads();
  for (int off = 64; off > 0; off >>= 1) {
    if (tid < off) { rs[tid] += rs[tid + off]; rs2[tid] += rs2[tid + off]; }
    __syncthreads();
  }
  float mu = rs[0] * (1.f / (float)CDIM);
  float var = rs2[0] * (1.f / (float)CDIM) - mu * mu;
  float rinv = rsqrtf(var + 1e-5f);
#pragma unroll
  for (int i = 0; i < 4; ++i) {
    int c = tid + i * 128;
    float o = shortcut[(size_t)row * CDIM + c] + (vals[i] - mu) * rinv * g[c] + bb[c];
    out[(size_t)row * CDIM + c] = o;
    if (out_bf) out_bf[(size_t)row * CDIM + c] = f2bf(o);
  }
}

// ---------------------------------------------------------------------------
// Host launch
// ---------------------------------------------------------------------------
extern "C" void kernel_launch(void* const* d_in, const int* in_sizes, int n_in,
                              void* d_out, int out_size, void* d_ws, size_t ws_size,
                              hipStream_t stream) {
  (void)in_sizes; (void)n_in; (void)out_size; (void)ws_size;
  const float*         x      = (const float*)d_in[0];
  const unsigned char* mask   = (const unsigned char*)d_in[1];
  const float* qkv_w  = (const float*)d_in[2];
  const float* q_bias = (const float*)d_in[3];
  const float* v_bias = (const float*)d_in[4];
  const float* lscale = (const float*)d_in[5];
  const float* cpb_w1 = (const float*)d_in[6];
  const float* cpb_b1 = (const float*)d_in[7];
  const float* cpb_w2 = (const float*)d_in[8];
  const float* proj_w = (const float*)d_in[9];
  const float* proj_b = (const float*)d_in[10];
  const float* n1g = (const float*)d_in[11];
  const float* n1b = (const float*)d_in[12];
  const float* n2g = (const float*)d_in[13];
  const float* n2b = (const float*)d_in[14];
  const float* fc1_w = (const float*)d_in[15];
  const float* fc1_b = (const float*)d_in[16];
  const float* fc2_w = (const float*)d_in[17];
  const float* fc2_b = (const float*)d_in[18];

  char* wp = (char*)d_ws;
  auto alloc = [&](size_t bytes) -> char* {
    char* p = wp;
    wp += (bytes + 255) & ~(size_t)255;
    return p;
  };
  unsigned short* wQKVW  = (unsigned short*)alloc((size_t)3 * CDIM * CDIM * 2);
  unsigned short* wPROJW = (unsigned short*)alloc((size_t)CDIM * CDIM * 2);
  unsigned short* wFC1W  = (unsigned short*)alloc((size_t)HIDDEN * CDIM * 2);
  unsigned short* wFC2W  = (unsigned short*)alloc((size_t)CDIM * HIDDEN * 2);
  unsigned short* wXW    = (unsigned short*)alloc((size_t)MROWS * CDIM * 2);
  unsigned short* wQ     = (unsigned short*)alloc((size_t)MROWS * CDIM * 2);
  unsigned short* wK     = (unsigned short*)alloc((size_t)MROWS * CDIM * 2);
  unsigned short* wV     = (unsigned short*)alloc((size_t)MROWS * CDIM * 2);
  float*          wMW    = (float*)alloc((size_t)MROWS * 4);
  float*          wTBL   = (float*)alloc((size_t)127 * HEADS * 4);
  float*          wBIAS  = (float*)alloc((size_t)HEADS * WIN * WIN * 4);
  unsigned short* wATTN  = (unsigned short*)alloc((size_t)MROWS * CDIM * 2);
  float*          wPROJO = (float*)alloc((size_t)MROWS * CDIM * 4);
  float*          wX3    = (float*)alloc((size_t)MROWS * CDIM * 4);
  unsigned short* wX3B   = (unsigned short*)alloc((size_t)MROWS * CDIM * 2);
  unsigned short* wH     = (unsigned short*)alloc((size_t)MROWS * HIDDEN * 2);
  float*          wMLP   = (float*)alloc((size_t)MROWS * CDIM * 4);

  // 1) weights -> bf16
  f2bf_kernel<<<(3 * CDIM * CDIM + 255) / 256, 256, 0, stream>>>(qkv_w, wQKVW, 3 * CDIM * CDIM);
  f2bf_kernel<<<(CDIM * CDIM + 255) / 256, 256, 0, stream>>>(proj_w, wPROJW, CDIM * CDIM);
  f2bf_kernel<<<(HIDDEN * CDIM + 255) / 256, 256, 0, stream>>>(fc1_w, wFC1W, HIDDEN * CDIM);
  f2bf_kernel<<<(CDIM * HIDDEN + 255) / 256, 256, 0, stream>>>(fc2_w, wFC2W, CDIM * HIDDEN);

  // 2) shifted windows + masks (also writes new_mask to d_out tail)
  prep_x_kernel<<<(MROWS * CDIM) / 256, 256, 0, stream>>>(x, wXW);
  masks_kernel<<<MROWS / 256, 256, 0, stream>>>(mask, wMW,
                                                (float*)d_out + (size_t)MROWS * CDIM);

  // 3) relative position bias MLP
  cpb_table_kernel<<<8, 128, 0, stream>>>(cpb_w1, cpb_b1, cpb_w2, wTBL);
  cpb_bias_kernel<<<(HEADS * WIN * WIN) / 256, 256, 0, stream>>>(wTBL, wBIAS);

  // 4) QKV GEMM (M=65536, N=1536, K=512) with per-head scatter
  {
    dim3 g((3 * CDIM) / 64, MROWS / 128);
    gemm_kernel<<<g, 256, 0, stream>>>(wXW, wQKVW, q_bias, v_bias,
                                       MROWS, 3 * CDIM, CDIM, 0, wQ, wK, wV, nullptr);
  }

  // 5) windowed cosine attention
  attn_kernel<<<NWIN * HEADS, 128, 0, stream>>>(wQ, wK, wV, wBIAS, wMW, lscale, wATTN);

  // 6) output projection (f32)
  {
    dim3 g(CDIM / 64, MROWS / 128);
    gemm_kernel<<<g, 256, 0, stream>>>(wATTN, wPROJW, proj_b, nullptr,
                                       MROWS, CDIM, CDIM, 1, nullptr, nullptr, nullptr, wPROJO);
  }

  // 7) unshift + LN1 + residual  -> x3 (f32 + bf16)
  ln_kernel<<<MROWS, 128, 0, stream>>>(wPROJO, x, n1g, n1b, 1, wX3, wX3B);

  // 8) FC1 + GELU (bf16 out)
  {
    dim3 g(HIDDEN / 64, MROWS / 128);
    gemm_kernel<<<g, 256, 0, stream>>>(wX3B, wFC1W, fc1_b, nullptr,
                                       MROWS, HIDDEN, CDIM, 2, wH, nullptr, nullptr, nullptr);
  }

  // 9) FC2 (f32 out)
  {
    dim3 g(CDIM / 64, MROWS / 128);
    gemm_kernel<<<g, 256, 0, stream>>>(wH, wFC2W, fc2_b, nullptr,
                                       MROWS, CDIM, HIDDEN, 1, nullptr, nullptr, nullptr, wMLP);
  }

  // 10) LN2 + residual -> d_out
  ln_kernel<<<MROWS, 128, 0, stream>>>(wMLP, wX3, n2g, n2b, 0, (float*)d_out, nullptr);
}
